// SCTConv_39195871544108
// MI455X (gfx1250) — matrix operand order repaired and checked by reading, not compiled, not run on hardware
//
#include <hip/hip_runtime.h>
#include <hip/hip_bf16.h>
#include <math.h>

// ---------------------------------------------------------------------------
// Vector types
// ---------------------------------------------------------------------------
typedef __attribute__((ext_vector_type(4)))  float          f32x4;
typedef __attribute__((ext_vector_type(8)))  float          v8f;
typedef __attribute__((ext_vector_type(2)))  unsigned int   u32x2;
typedef __attribute__((ext_vector_type(4)))  unsigned int   u32x4;
typedef __attribute__((ext_vector_type(4)))  unsigned short u16x4;
typedef __attribute__((ext_vector_type(8)))  unsigned short u16x8;
typedef __attribute__((ext_vector_type(16))) unsigned short u16x16;
typedef __attribute__((ext_vector_type(16))) __bf16         v16bf;

__device__ __forceinline__ unsigned short f2bf(float f) {
    unsigned int u = __builtin_bit_cast(unsigned int, f);
    return (unsigned short)((u + 0x7FFFu + ((u >> 16) & 1u)) >> 16);
}

// Pack two floats into two bf16 (RNE). Uses v_cvt_pk_bf16_f32 when available.
__device__ __forceinline__ unsigned int f2bf2(float a, float b) {
#if __has_builtin(__builtin_amdgcn_cvt_pk_bf16_f32)
    typedef __attribute__((ext_vector_type(2))) __bf16 bf16x2;
    bf16x2 p = __builtin_amdgcn_cvt_pk_bf16_f32(a, b);
    return __builtin_bit_cast(unsigned int, p);
#else
    return (unsigned int)f2bf(a) | ((unsigned int)f2bf(b) << 16);
#endif
}

__device__ __forceinline__ float leaky01(float v) {
    return v >= 0.0f ? v : 0.01f * v;
}

// Build a v16bf fragment from two contiguous 8-element (16B) LDS chunks.
__device__ __forceinline__ v16bf mk_frag(const unsigned short* p0,
                                         const unsigned short* p1) {
    u16x8 lo = *(const u16x8*)p0;
    u16x8 hi = *(const u16x8*)p1;
    u16x16 c = __builtin_shufflevector(lo, hi, 0, 1, 2, 3, 4, 5, 6, 7,
                                       8, 9, 10, 11, 12, 13, 14, 15);
    return __builtin_bit_cast(v16bf, c);
}

// ---------------------------------------------------------------------------
// Degree kernels (deterministic: partial sums + fixed-order reduction)
// ---------------------------------------------------------------------------
__global__ void colsum_partial(const float* __restrict__ adj,
                               float* __restrict__ cspart, int N) {
    int j = blockIdx.x * blockDim.x + threadIdx.x;
    int chunk = blockIdx.y;
    int rows = N / gridDim.y;
    int r0 = chunk * rows;
    float s = 0.0f;
    for (int i = 0; i < rows; ++i) s += adj[(size_t)(r0 + i) * N + j];
    cspart[(size_t)chunk * N + j] = s;
}

__global__ void finalize_deg(const float* __restrict__ cspart,
                             float* __restrict__ dg,
                             float* __restrict__ dinv, int N, int nch) {
    int j = blockIdx.x * blockDim.x + threadIdx.x;
    if (j >= N) return;
    float s = 0.0f;
    for (int c = 0; c < nch; ++c) s += cspart[(size_t)c * N + j];
    dg[j]   = rsqrtf(s + 1.0f);   // colsum(adj + I) = colsum(adj) + 1
    dinv[j] = 1.0f / s;
}

// ---------------------------------------------------------------------------
// Elementwise prep / post kernels
// ---------------------------------------------------------------------------
__global__ void prep_gcn(const float* __restrict__ x, const float* __restrict__ dg,
                         float* __restrict__ uf32, unsigned short* __restrict__ ubf,
                         size_t NH, int H) {
    size_t idx = (size_t)blockIdx.x * blockDim.x + threadIdx.x;
    if (idx >= NH) return;
    float u = dg[idx / H] * x[idx];
    uf32[idx] = u;
    ubf[idx]  = f2bf(u);
}

__global__ void post_gcn(const float* __restrict__ y, const float* __restrict__ uf32,
                         const float* __restrict__ dg, float* __restrict__ xout,
                         float* __restrict__ hout, size_t NH, int H) {
    size_t idx = (size_t)blockIdx.x * blockDim.x + threadIdx.x;
    if (idx >= NH) return;
    // x_new = D * (A_gcn @ (D x)) = D * (adj @ u + u)
    float xn = dg[idx / H] * (y[idx] + uf32[idx]);
    xout[idx] = xn;
    hout[idx] = leaky01(xn);
}

__global__ void prep_sct(const float* __restrict__ fp, const float* __restrict__ dinv,
                         unsigned short* __restrict__ ubf, size_t NH, int H) {
    size_t idx = (size_t)blockIdx.x * blockDim.x + threadIdx.x;
    if (idx >= NH) return;
    ubf[idx] = f2bf(dinv[idx / H] * fp[idx]);
}

__global__ void post_sct(const float* __restrict__ fp_in, const float* __restrict__ y,
                         float* __restrict__ fp_out, size_t NH) {
    size_t idx = (size_t)blockIdx.x * blockDim.x + threadIdx.x;
    if (idx >= NH) return;
    fp_out[idx] = 0.5f * fp_in[idx] + 0.5f * y[idx];
}

__global__ void snap_copy(const float* __restrict__ fp, float* __restrict__ prev,
                          size_t NH) {
    size_t idx = (size_t)blockIdx.x * blockDim.x + threadIdx.x;
    if (idx >= NH) return;
    prev[idx] = fp[idx];
}

__global__ void snap_diff(const float* __restrict__ fp, float* __restrict__ prev,
                          const int* __restrict__ moment,
                          float* __restrict__ hout, size_t NH) {
    size_t idx = (size_t)blockIdx.x * blockDim.x + threadIdx.x;
    if (idx >= NH) return;
    int m = moment[0];
    float d = fabsf(prev[idx] - fp[idx]);
    hout[idx] = (m == 1) ? d : powf(d, (float)m);
    prev[idx] = fp[idx];
}

// ---------------------------------------------------------------------------
// Main GEMM:  C[M x 256] = A(f32, MxK) @ B(bf16, Kx256)
// Block tile 64x256, BK=32, 8 waves (2x4); wave tile 32x64:
//   2 A-frags x 4 B-frags -> 8 v_wmma_f32_16x16x32_bf16 per K-stage.
// A converted f32->bf16 on the LDS store; B staged transposed [n][k].
// All fragments loaded before the WMMA burst so ds_loads overlap.
// ---------------------------------------------------------------------------
__global__ __launch_bounds__(256) void gemm_adj_kernel(
    const float* __restrict__ A, const unsigned short* __restrict__ B,
    float* __restrict__ C, int M, int K) {
    constexpr int BK = 32, BM = 64, BN = 256, LDT = BK + 8;
    __shared__ __align__(16) unsigned short As[2][BM][LDT];
    __shared__ __align__(16) unsigned short Bs[2][BN][LDT];

    const int tid  = threadIdx.x;
    const int lane = tid & 31;
    const int wave = tid >> 5;
    const int m0   = blockIdx.x * BM;

    // A load mapping: 8 f32 per thread (two f32x4)
    const int arow = tid >> 2;          // 0..63
    const int acol = (tid & 3) << 3;    // 0,8,16,24
    // B load mapping: 4 k-rows x 8 n per thread
    const int bkq = tid >> 5;
    const int bn0 = (tid & 31) << 3;

    const int nsteps = K / BK;

    f32x4 aR0, aR1;
    u16x8 bReg[4];

    // ---- stage 0 global loads ----
    {
        const float* ap = A + (size_t)(m0 + arow) * K + acol;
        aR0 = *(const f32x4*)ap;
        aR1 = *(const f32x4*)(ap + 4);
    }
#pragma unroll
    for (int i = 0; i < 4; ++i)
        bReg[i] = *(const u16x8*)(B + (size_t)(bkq * 4 + i) * BN + bn0);
    // ---- stage 0 LDS stores ----
    {
        u32x4 pk = {f2bf2(aR0[0], aR0[1]), f2bf2(aR0[2], aR0[3]),
                    f2bf2(aR1[0], aR1[1]), f2bf2(aR1[2], aR1[3])};
        *(u32x4*)&As[0][arow][acol] = pk;
#pragma unroll
        for (int i = 0; i < 4; ++i)
#pragma unroll
            for (int j = 0; j < 8; ++j) Bs[0][bn0 + j][bkq * 4 + i] = bReg[i][j];
    }
    __syncthreads();

    const int wm = (wave >> 2) * 32;   // 0 or 32
    const int wn = (wave & 3) * 64;    // 0,64,128,192
    const int kg = lane >> 4;
    const int ml = lane & 15;

    v8f acc[2][4];
#pragma unroll
    for (int i = 0; i < 2; ++i)
#pragma unroll
        for (int j = 0; j < 4; ++j)
#pragma unroll
            for (int v = 0; v < 8; ++v) acc[i][j][v] = 0.0f;

    for (int kt = 0; kt < nsteps; ++kt) {
        const int buf = kt & 1;
        const bool more = (kt + 1) < nsteps;
        if (more) {
            const int k0 = (kt + 1) * BK;
            const float* ap = A + (size_t)(m0 + arow) * K + k0 + acol;
            aR0 = *(const f32x4*)ap;
            aR1 = *(const f32x4*)(ap + 4);
#pragma unroll
            for (int i = 0; i < 4; ++i)
                bReg[i] = *(const u16x8*)(B + (size_t)(k0 + bkq * 4 + i) * BN + bn0);
        }
        // ---- load ALL fragments, then burst the 8 WMMAs ----
        {
            const unsigned short* arp0 = &As[buf][wm + ml][0];
            const unsigned short* arp1 = &As[buf][wm + 16 + ml][0];
            v16bf av0 = mk_frag(arp0 + 8 * kg, arp0 + 16 + 8 * kg);
            v16bf av1 = mk_frag(arp1 + 8 * kg, arp1 + 16 + 8 * kg);
            v16bf bvf[4];
#pragma unroll
            for (int j = 0; j < 4; ++j) {
                const unsigned short* brp = &Bs[buf][wn + j * 16 + ml][0];
                bvf[j] = mk_frag(brp + 8 * kg, brp + 16 + 8 * kg);
            }
#pragma unroll
            for (int j = 0; j < 4; ++j) {
                acc[0][j] = __builtin_amdgcn_wmma_f32_16x16x32_bf16(
                    false, av0, false, bvf[j], (short)0, acc[0][j], false, false);
                acc[1][j] = __builtin_amdgcn_wmma_f32_16x16x32_bf16(
                    false, av1, false, bvf[j], (short)0, acc[1][j], false, false);
            }
        }
        if (more) {
            const int nb = buf ^ 1;
            u32x4 pk = {f2bf2(aR0[0], aR0[1]), f2bf2(aR0[2], aR0[3]),
                        f2bf2(aR1[0], aR1[1]), f2bf2(aR1[2], aR1[3])};
            *(u32x4*)&As[nb][arow][acol] = pk;
#pragma unroll
            for (int i = 0; i < 4; ++i)
#pragma unroll
                for (int j = 0; j < 8; ++j)
                    Bs[nb][bn0 + j][bkq * 4 + i] = bReg[i][j];
        }
        __syncthreads();
    }

    // C/D layout: lanes 0-15 -> M = v, lanes 16-31 -> M = v+8, N = lane&15
#pragma unroll
    for (int i = 0; i < 2; ++i) {
        const int mbase = m0 + wm + i * 16 + ((lane < 16) ? 0 : 8);
#pragma unroll
        for (int j = 0; j < 4; ++j) {
            const int n = wn + j * 16 + ml;
#pragma unroll
            for (int v = 0; v < 8; ++v)
                C[(size_t)(mbase + v) * BN + n] = acc[i][j][v];
        }
    }
}

// ---------------------------------------------------------------------------
// MLP GEMM:  out = leaky(A(bf16, Mx256) @ W^T + bias); W is [256][256] f32.
// B-matrix element (k, n) = W[n][k], so LDS tile Bs[n][k] = W[n][k].
// Same 64x256 block / 32x64 wave tiling as the adj GEMM.
// ---------------------------------------------------------------------------
__global__ __launch_bounds__(256) void gemm_mlp_kernel(
    const unsigned short* __restrict__ A, const float* __restrict__ W,
    const float* __restrict__ bias, float* __restrict__ Cf,
    unsigned short* __restrict__ Cb, int M) {
    constexpr int BK = 32, BM = 64, BN = 256, K = 256, LDT = BK + 8;
    __shared__ __align__(16) unsigned short As[2][BM][LDT];
    __shared__ __align__(16) unsigned short Bs[2][BN][LDT];

    const int tid  = threadIdx.x;
    const int lane = tid & 31;
    const int wave = tid >> 5;
    const int m0   = blockIdx.x * BM;

    const int arow = tid >> 2;          // 0..63
    const int acol = (tid & 3) << 3;    // 0,8,16,24 (ushorts)
    const int bnq  = tid >> 3;          // 0..31
    const int bcol = (tid & 7) << 2;    // 0..28

    const int nsteps = K / BK;

    u16x8 aReg;
    f32x4 wReg[8];

    aReg = *(const u16x8*)(A + (size_t)(m0 + arow) * K + acol);
#pragma unroll
    for (int r = 0; r < 8; ++r)
        wReg[r] = *(const f32x4*)(W + (size_t)(bnq + r * 32) * K + bcol);
    {
        *(u16x8*)&As[0][arow][acol] = aReg;
#pragma unroll
        for (int r = 0; r < 8; ++r) {
            u32x2 bv = {f2bf2(wReg[r][0], wReg[r][1]), f2bf2(wReg[r][2], wReg[r][3])};
            *(u32x2*)&Bs[0][bnq + r * 32][bcol] = bv;
        }
    }
    __syncthreads();

    const int wm = (wave >> 2) * 32;
    const int wn = (wave & 3) * 64;
    const int kg = lane >> 4;
    const int ml = lane & 15;

    v8f acc[2][4];
#pragma unroll
    for (int i = 0; i < 2; ++i)
#pragma unroll
        for (int j = 0; j < 4; ++j)
#pragma unroll
            for (int v = 0; v < 8; ++v) acc[i][j][v] = 0.0f;

    for (int kt = 0; kt < nsteps; ++kt) {
        const int buf = kt & 1;
        const bool more = (kt + 1) < nsteps;
        if (more) {
            const int k0 = (kt + 1) * BK;
            aReg = *(const u16x8*)(A + (size_t)(m0 + arow) * K + k0 + acol);
#pragma unroll
            for (int r = 0; r < 8; ++r)
                wReg[r] = *(const f32x4*)(W + (size_t)(bnq + r * 32) * K + k0 + bcol);
        }
        {
            const unsigned short* arp0 = &As[buf][wm + ml][0];
            const unsigned short* arp1 = &As[buf][wm + 16 + ml][0];
            v16bf av0 = mk_frag(arp0 + 8 * kg, arp0 + 16 + 8 * kg);
            v16bf av1 = mk_frag(arp1 + 8 * kg, arp1 + 16 + 8 * kg);
            v16bf bvf[4];
#pragma unroll
            for (int j = 0; j < 4; ++j) {
                const unsigned short* brp = &Bs[buf][wn + j * 16 + ml][0];
                bvf[j] = mk_frag(brp + 8 * kg, brp + 16 + 8 * kg);
            }
#pragma unroll
            for (int j = 0; j < 4; ++j) {
                acc[0][j] = __builtin_amdgcn_wmma_f32_16x16x32_bf16(
                    false, av0, false, bvf[j], (short)0, acc[0][j], false, false);
                acc[1][j] = __builtin_amdgcn_wmma_f32_16x16x32_bf16(
                    false, av1, false, bvf[j], (short)0, acc[1][j], false, false);
            }
        }
        if (more) {
            const int nb = buf ^ 1;
            *(u16x8*)&As[nb][arow][acol] = aReg;
#pragma unroll
            for (int r = 0; r < 8; ++r) {
                u32x2 bv = {f2bf2(wReg[r][0], wReg[r][1]),
                            f2bf2(wReg[r][2], wReg[r][3])};
                *(u32x2*)&Bs[nb][bnq + r * 32][bcol] = bv;
            }
        }
        __syncthreads();
    }

#pragma unroll
    for (int i = 0; i < 2; ++i) {
        const int mbase = m0 + wm + i * 16 + ((lane < 16) ? 0 : 8);
#pragma unroll
        for (int j = 0; j < 4; ++j) {
            const int n = wn + j * 16 + ml;
            const float bv = bias[n];
#pragma unroll
            for (int v = 0; v < 8; ++v) {
                float val = leaky01(acc[i][j][v] + bv);
                size_t idx = (size_t)(mbase + v) * BN + n;
                if (Cf) Cf[idx] = val;
                if (Cb) Cb[idx] = f2bf(val);
            }
        }
    }
}

// ---------------------------------------------------------------------------
// Attention over 6 channels: one block (256 threads == H) per node.
// e[n,s] = sum_k relu(X[n,k])*a[k] + sum_k relu(h[s][n,k])*a[H+k]
// h_prime = (1/6) * sum_s softmax(e)_s * h[s]   -> stored as bf16 for the MLP
// ---------------------------------------------------------------------------
__global__ __launch_bounds__(256) void attention_kernel(
    const float* __restrict__ X, const float* __restrict__ a,
    const float* __restrict__ hbuf, unsigned short* __restrict__ ubf,
    size_t NH) {
    __shared__ float red[256];
    const int n = blockIdx.x;
    const int k = threadIdx.x;
    const size_t base = (size_t)n * 256 + k;

    const float a1 = a[k];
    const float a2 = a[256 + k];
    const float xk = X[base];

    red[k] = fmaxf(xk, 0.0f) * a1;
    __syncthreads();
#pragma unroll
    for (int s = 128; s > 0; s >>= 1) {
        if (k < s) red[k] += red[k + s];
        __syncthreads();
    }
    const float c0 = red[0];
    __syncthreads();

    float hv[6], e[6];
#pragma unroll
    for (int s = 0; s < 6; ++s) {
        hv[s] = hbuf[(size_t)s * NH + base];
        red[k] = fmaxf(hv[s], 0.0f) * a2;
        __syncthreads();
#pragma unroll
        for (int t = 128; t > 0; t >>= 1) {
            if (k < t) red[k] += red[k + t];
            __syncthreads();
        }
        e[s] = c0 + red[0];
        __syncthreads();
    }

    float emax = e[0];
#pragma unroll
    for (int s = 1; s < 6; ++s) emax = fmaxf(emax, e[s]);
    float w[6], wsum = 0.0f;
#pragma unroll
    for (int s = 0; s < 6; ++s) {
        w[s] = expf(e[s] - emax);
        wsum += w[s];
    }
    const float inv = 1.0f / (6.0f * wsum);
    float hp = 0.0f;
#pragma unroll
    for (int s = 0; s < 6; ++s) hp += w[s] * hv[s];
    ubf[base] = f2bf(hp * inv);
}

// ---------------------------------------------------------------------------
// Host orchestration
// ---------------------------------------------------------------------------
extern "C" void kernel_launch(void* const* d_in, const int* in_sizes, int n_in,
                              void* d_out, int out_size, void* d_ws, size_t ws_size,
                              hipStream_t stream) {
    (void)n_in; (void)out_size; (void)ws_size;
    const float* X     = (const float*)d_in[0];
    const float* adj   = (const float*)d_in[1];
    const float* avec  = (const float*)d_in[2];
    const float* W1    = (const float*)d_in[3];
    const float* b1    = (const float*)d_in[4];
    const float* W2    = (const float*)d_in[5];
    const float* b2    = (const float*)d_in[6];
    const int*   mom   = (const int*)d_in[7];

    const int H = in_sizes[4];          // 256
    const int N = in_sizes[0] / H;      // 8192
    const size_t NH = (size_t)N * H;

    char* ws = (char*)d_ws;
    size_t off = 0;
    auto wsalloc = [&](size_t bytes) -> char* {
        char* p = ws + off;
        off += (bytes + 255) & ~(size_t)255;
        return p;
    };
    constexpr int RCH = 64;
    float* cspart = (float*)wsalloc((size_t)RCH * N * 4);
    float* dg     = (float*)wsalloc((size_t)N * 4);
    float* dinv   = (float*)wsalloc((size_t)N * 4);
    unsigned short* ubf  = (unsigned short*)wsalloc(NH * 2);
    unsigned short* h1bf = (unsigned short*)wsalloc(NH * 2);
    float* uf32  = (float*)wsalloc(NH * 4);
    float* xbuf  = (float*)wsalloc(NH * 4);
    float* ybuf  = (float*)wsalloc(NH * 4);
    float* fpbuf = (float*)wsalloc(NH * 4);
    float* prev  = (float*)wsalloc(NH * 4);
    float* hbuf  = (float*)wsalloc(6 * NH * 4);

    const dim3 ew((unsigned)((NH + 255) / 256));
    const dim3 gemmGrid((unsigned)(N / 64));

    // degree vectors
    colsum_partial<<<dim3(N / 256, RCH), 256, 0, stream>>>(adj, cspart, N);
    finalize_deg<<<(N + 255) / 256, 256, 0, stream>>>(cspart, dg, dinv, N, RCH);

    // ---- GCN diffusion, 3 hops ----
    const float* xc = X;
    for (int t = 0; t < 3; ++t) {
        prep_gcn<<<ew, 256, 0, stream>>>(xc, dg, uf32, ubf, NH, H);
        gemm_adj_kernel<<<gemmGrid, 256, 0, stream>>>(adj, ubf, ybuf, N, N);
        post_gcn<<<ew, 256, 0, stream>>>(ybuf, uf32, dg, xbuf,
                                         hbuf + (size_t)t * NH, NH, H);
        xc = xbuf;
    }

    // ---- Scattering diffusion, 8 steps, snapshots at {0,1,3,7} ----
    const float* fc = X;
    int hslot = 3;
    for (int i = 0; i < 8; ++i) {
        prep_sct<<<ew, 256, 0, stream>>>(fc, dinv, ubf, NH, H);
        gemm_adj_kernel<<<gemmGrid, 256, 0, stream>>>(adj, ubf, ybuf, N, N);
        post_sct<<<ew, 256, 0, stream>>>(fc, ybuf, fpbuf, NH);
        fc = fpbuf;
        if (i == 0) {
            snap_copy<<<ew, 256, 0, stream>>>(fpbuf, prev, NH);
        } else if (i == 1 || i == 3 || i == 7) {
            snap_diff<<<ew, 256, 0, stream>>>(fpbuf, prev, mom,
                                              hbuf + (size_t)hslot * NH, NH);
            ++hslot;
        }
    }

    // ---- attention over the 6 channels -> h_prime (bf16 in ubf) ----
    attention_kernel<<<N, 256, 0, stream>>>(X, avec, hbuf, ubf, NH);

    // ---- output MLP ----
    gemm_mlp_kernel<<<gemmGrid, 256, 0, stream>>>(ubf, W1, b1, nullptr, h1bf, N);
    gemm_mlp_kernel<<<gemmGrid, 256, 0, stream>>>(h1bf, W2, b2, (float*)d_out,
                                                  nullptr, N);
}
